// ResNetBase_68350109548653
// MI455X (gfx1250) — matrix operand order repaired and checked by reading, not compiled, not run on hardware
//
#include <hip/hip_runtime.h>
#include <hip/hip_bf16.h>
#include <stdint.h>

// ---------------------------------------------------------------------------
// Types for CDNA5 WMMA (gfx1250, wave32)
// ---------------------------------------------------------------------------
typedef __attribute__((ext_vector_type(16))) __bf16    v16bf;
typedef __attribute__((ext_vector_type(8)))  float     v8f;
typedef __attribute__((ext_vector_type(8)))  uint16_t  us8;

union FragAB { v16bf v; us8 h[2]; };
union Acc    { v8f   v; float f[8]; };

__device__ __forceinline__ uint16_t f2bf(float f) {
  uint32_t u = __builtin_bit_cast(uint32_t, f);
  uint32_t r = u + 0x7fffu + ((u >> 16) & 1u);   // round-to-nearest-even
  return (uint16_t)(r >> 16);
}
__device__ __forceinline__ float bf2f(uint16_t h) {
  uint32_t u = ((uint32_t)h) << 16;
  return __builtin_bit_cast(float, u);
}

// ---------------------------------------------------------------------------
// Implicit-GEMM 3D convolution with v_wmma_f32_16x16x32_bf16
//   A: im2col of activations, NDHWC bf16 (LDS-staged, zero-padded borders)
//   B: weights [Cout][KP] bf16, staged via GLOBAL_LOAD_ASYNC_TO_LDS_B128
//   Block tile: 128(M) x 64(N), K stepped by 32, double-buffered LDS.
// ---------------------------------------------------------------------------
#define BM 128
#define BN 64
#define BK 32
#define APITCH 40   // elems; 80B row pitch -> 16B aligned, conflict-free
#define BPITCH 40

__global__ __launch_bounds__(256) void conv3d_wmma(
    const uint16_t* __restrict__ inp,  // [Nb,Di,Di,Di,Cin] bf16
    const uint16_t* __restrict__ wt,   // [Cout,KP] bf16
    float* __restrict__ out,           // [M,Cout] fp32
    int Nb, int Di, int Do, int Cin, int Cout,
    int kx, int stride, int pad, int K, int KP)
{
  __shared__ __align__(16) uint16_t ldsA[2][BM * APITCH];
  __shared__ __align__(16) uint16_t ldsB[2][BN * BPITCH];

  const int tid  = (int)threadIdx.x;
  const int wave = tid >> 5;
  const int lane = tid & 31;
  const int M    = Nb * Do * Do * Do;
  const int mBase = (int)blockIdx.x * BM;
  const int n0    = (int)blockIdx.y * BN;

  // ---- A staging assignment: 2 threads per row, 16 K-elems each ----
  const int ar = tid >> 1;
  const int ah = tid & 1;
  const int am = mBase + ar;
  const bool amv = am < M;
  int an = 0, az = 0, ay = 0, ax = 0;
  if (amv) {
    int t = am;
    ax = t % Do; t /= Do;
    ay = t % Do; t /= Do;
    az = t % Do; t /= Do;
    an = t;
  }
  const bool fast = (Cin & 15) == 0;   // 16-seg never crosses a Cin run

  // Incrementally-maintained im2col decomposition of kkA (fast path).
  // kkA starts at ah*16 < 32 <= Cin, so (ci,kw,kh,kd) = (kkA,0,0,0).
  int kkA = ah * 16;
  int ci = kkA, kw = 0, kh = 0, kd = 0;

  // ---- B staging assignment: one 16B async segment per lane ----
  const int bn  = tid >> 2;
  const int seg = tid & 3;
  int kkB = 0;
  const unsigned long long wbase = (unsigned long long)(uintptr_t)wt;

  auto stageA = [&](int buf) {
    uint16_t* dst = &ldsA[buf][ar * APITCH + ah * 16];
    bool wrote = false;
    if (amv && kkA < K) {
      if (fast) {
        const int zi = az * stride - pad + kd;
        const int yi = ay * stride - pad + kh;
        const int xi = ax * stride - pad + kw;
        if ((unsigned)zi < (unsigned)Di && (unsigned)yi < (unsigned)Di &&
            (unsigned)xi < (unsigned)Di) {
          const uint16_t* src =
              inp + ((size_t)(((an * Di + zi) * Di + yi) * Di + xi)) * Cin + ci;
          *(us8*)dst       = *(const us8*)src;
          *(us8*)(dst + 8) = *(const us8*)(src + 8);
          wrote = true;
        }
      } else {  // scalar decode (stem, Cin=3)
        for (int j = 0; j < 16; ++j) {
          const int kk = kkA + j;
          uint16_t v = 0;
          if (kk < K) {
            int cc = kk % Cin;
            int t  = kk / Cin;
            int w2 = t % kx; t /= kx;
            int h2 = t % kx;
            int d2 = t / kx;
            int zi = az * stride - pad + d2;
            int yi = ay * stride - pad + h2;
            int xi = ax * stride - pad + w2;
            if ((unsigned)zi < (unsigned)Di && (unsigned)yi < (unsigned)Di &&
                (unsigned)xi < (unsigned)Di)
              v = inp[((size_t)(((an * Di + zi) * Di + yi) * Di + xi)) * Cin + cc];
          }
          dst[j] = v;
        }
        wrote = true;
      }
    }
    if (!wrote) { *(us8*)dst = (us8)(0); *(us8*)(dst + 8) = (us8)(0); }
    // advance kkA by BK and update (ci,kw,kh,kd) incrementally (BK <= Cin here)
    kkA += BK;
    if (fast) {
      ci += BK;
      if (ci >= Cin) {
        ci -= Cin;
        if (++kw == kx) { kw = 0; if (++kh == kx) { kh = 0; ++kd; } }
      }
    }
  };

  auto stageB = [&](int buf) {
    // async DMA: global (GVS: sgpr base + vgpr byte offset) -> LDS, 16B/lane
    const uint32_t lds_dst =
        (uint32_t)(uintptr_t)&ldsB[buf][bn * BPITCH + seg * 8];
    const uint32_t voff =
        (uint32_t)(((size_t)(n0 + bn) * KP + kkB + seg * 8) * 2);
    asm volatile("global_load_async_to_lds_b128 %0, %1, %2"
                 :: "v"(lds_dst), "v"(voff), "s"(wbase)
                 : "memory");
    // warm GL2 two chunks ahead
    if (kkB + 2 * BK < KP)
      __builtin_prefetch(wt + (size_t)(n0 + bn) * KP + kkB + 2 * BK, 0, 1);
    kkB += BK;
  };

  auto waitAsync = [&]() {
    asm volatile("s_wait_asynccnt 0x0" ::: "memory");
  };

  Acc acc[4];
#pragma unroll
  for (int t = 0; t < 4; ++t) acc[t].v = (v8f)(0.0f);

  const int nch = KP / BK;
  stageA(0);
  stageB(0);
  waitAsync();
  __syncthreads();

  for (int i = 0; i < nch; ++i) {
    const int p = i & 1;
    if (i + 1 < nch) {  // stage next chunk while computing this one
      stageA(1 - p);
      stageB(1 - p);
    }
    // ---------------- fragments + WMMA ----------------
    FragAB a;
    {
      const int row = wave * 16 + (lane & 15);
      const int ko  = (lane >= 16) ? 8 : 0;        // ISA 16-bit A 16x32 layout
      a.h[0] = *(const us8*)&ldsA[p][row * APITCH + ko];
      a.h[1] = *(const us8*)&ldsA[p][row * APITCH + 16 + ko];
    }
#pragma unroll
    for (int t = 0; t < 4; ++t) {
      FragAB b;
      const int row = t * 16 + (lane & 15);
      const int ko  = (lane >= 16) ? 8 : 0;        // ISA 16-bit B 32x16 layout
      b.h[0] = *(const us8*)&ldsB[p][row * BPITCH + ko];
      b.h[1] = *(const us8*)&ldsB[p][row * BPITCH + 16 + ko];
      acc[t].v = __builtin_amdgcn_wmma_f32_16x16x32_bf16(
          false, a.v, false, b.v, (short)0, acc[t].v, false, false);
    }
    waitAsync();      // next-chunk async B load finished during compute
    __syncthreads();  // all waves done reading buf p / writing buf 1-p
  }

  // ---------------- epilogue: fp32 store ----------------
  const int mrow0 = mBase + wave * 16 + ((lane >= 16) ? 8 : 0);
  const int col0  = n0 + (lane & 15);
#pragma unroll
  for (int t = 0; t < 4; ++t) {
#pragma unroll
    for (int r = 0; r < 8; ++r) {
      const int m = mrow0 + r;
      if (m < M) out[(size_t)m * Cout + col0 + t * 16] = acc[t].f[r];
    }
  }
}

// ---------------------------------------------------------------------------
// Packing kernels
// ---------------------------------------------------------------------------
__global__ void pack_input_kernel(const float* __restrict__ x,
                                  uint16_t* __restrict__ out,
                                  int Nb, int C, int D) {
  long long total = (long long)Nb * C * D * D * D;
  long long i = (long long)blockIdx.x * blockDim.x + threadIdx.x;
  if (i >= total) return;
  long long t = i;
  int xw = (int)(t % D); t /= D;
  int y  = (int)(t % D); t /= D;
  int z  = (int)(t % D); t /= D;
  int c  = (int)(t % C);
  int n  = (int)(t / C);
  out[((long long)((n * D + z) * D + y) * D + xw) * C + c] = f2bf(x[i]);
}

__global__ void pack_weight_kernel(const float* __restrict__ w,  // OIDHW
                                   uint16_t* __restrict__ wt,    // [Cout][KP]
                                   int Cin, int kx, int K, int KP,
                                   long long total) {
  long long i = (long long)blockIdx.x * blockDim.x + threadIdx.x;
  if (i >= total) return;
  int k = (int)(i % KP);
  long long o = i / KP;
  uint16_t v = 0;
  if (k < K) {
    int ci = k % Cin;
    int t  = k / Cin;
    int kw = t % kx; t /= kx;
    int kh = t % kx;
    int kd = t / kx;
    v = f2bf(w[(((o * Cin + ci) * kx + kd) * kx + kh) * kx + kw]);
  }
  wt[i] = v;
}

// ---------------------------------------------------------------------------
// Norm stats (BN: groups=1; InstanceNorm: groups=Nb), fused norm+act+residual
// ---------------------------------------------------------------------------
__global__ __launch_bounds__(256) void stats_kernel(
    const float* __restrict__ x, const float* __restrict__ g,
    const float* __restrict__ b, float* __restrict__ scale,
    float* __restrict__ bias, int S, int C) {
  __shared__ float s1[256], s2[256];
  const int c   = (int)blockIdx.x;
  const int grp = (int)blockIdx.y;
  const float* base = x + (size_t)grp * S * C + c;
  float sum = 0.f, sq = 0.f;
  for (int i = (int)threadIdx.x; i < S; i += 256) {
    float v = base[(size_t)i * C];
    sum += v; sq += v * v;
  }
  s1[threadIdx.x] = sum; s2[threadIdx.x] = sq;
  __syncthreads();
  for (int o = 128; o > 0; o >>= 1) {
    if ((int)threadIdx.x < o) {
      s1[threadIdx.x] += s1[threadIdx.x + o];
      s2[threadIdx.x] += s2[threadIdx.x + o];
    }
    __syncthreads();
  }
  if (threadIdx.x == 0) {
    float m   = s1[0] / (float)S;
    float var = s2[0] / (float)S - m * m;
    float gg  = g ? g[c] : 1.f;
    float bb  = b ? b[c] : 0.f;
    float sc  = gg * rsqrtf(var + 1e-5f);
    scale[grp * C + c] = sc;
    bias[grp * C + c]  = bb - m * sc;
  }
}

__global__ void norm_act_kernel(const float* __restrict__ x,
                                const float* __restrict__ scale,
                                const float* __restrict__ bias,
                                const uint16_t* __restrict__ res,
                                uint16_t* __restrict__ out,
                                long long total, int C, int S, int act) {
  long long i = (long long)blockIdx.x * blockDim.x + threadIdx.x;
  if (i >= total) return;
  int c = (int)(i % C);
  long long r = i / C;
  int grp = S ? (int)(r / S) : 0;
  float v = x[i] * scale[grp * C + c] + bias[grp * C + c];
  if (res) v += bf2f(res[i]);
  if (act == 1) {
    v = fmaxf(v, 0.f);
  } else if (act == 2) {  // tanh-approx GELU (jax.nn.gelu default)
    float t = 0.7978845608028654f * (v + 0.044715f * v * v * v);
    v = 0.5f * v * (1.f + tanhf(t));
  }
  out[i] = f2bf(v);
}

// ---------------------------------------------------------------------------
// 2x2x2 max pool (SAME, even dims -> exact), NDHWC bf16
// ---------------------------------------------------------------------------
__global__ void maxpool2_kernel(const uint16_t* __restrict__ in,
                                uint16_t* __restrict__ out,
                                int Nb, int Di, int C) {
  const int Do = Di >> 1;
  long long total = (long long)Nb * Do * Do * Do * C;
  long long i = (long long)blockIdx.x * blockDim.x + threadIdx.x;
  if (i >= total) return;
  long long t = i;
  int c = (int)(t % C); t /= C;
  int x = (int)(t % Do); t /= Do;
  int y = (int)(t % Do); t /= Do;
  int z = (int)(t % Do);
  int n = (int)(t / Do);
  float m = -3.4e38f;
  for (int dz = 0; dz < 2; ++dz)
    for (int dy = 0; dy < 2; ++dy)
      for (int dx = 0; dx < 2; ++dx) {
        long long idx =
            ((long long)(((n * Di + 2 * z + dz) * Di + 2 * y + dy) * Di +
                         2 * x + dx)) * C + c;
        m = fmaxf(m, bf2f(in[idx]));
      }
  out[i] = f2bf(m);
}

// ---------------------------------------------------------------------------
// Global max pool + final linear head  ->  out [Nb, NC] fp32
// ---------------------------------------------------------------------------
__global__ void head_kernel(const uint16_t* __restrict__ x,  // [Nb,S,C] bf16
                            const float* __restrict__ W,     // [C,NC]
                            const float* __restrict__ bvec,  // [NC]
                            float* __restrict__ out, int Nb, int S, int C,
                            int NC) {
  int i = (int)(blockIdx.x * blockDim.x + threadIdx.x);
  if (i >= Nb * NC) return;
  int n = i / NC, j = i % NC;
  float accum = bvec[j];
  for (int c = 0; c < C; ++c) {
    float m = -3.4e38f;
    for (int s = 0; s < S; ++s)
      m = fmaxf(m, bf2f(x[((size_t)n * S + s) * C + c]));
    accum += m * W[c * NC + j];
  }
  out[i] = accum;
}

// ---------------------------------------------------------------------------
// Host orchestration
// ---------------------------------------------------------------------------
extern "C" void kernel_launch(void* const* d_in, const int* in_sizes, int n_in,
                              void* d_out, int out_size, void* d_ws,
                              size_t ws_size, hipStream_t stream) {
  (void)in_sizes; (void)out_size; (void)ws_size;
  if (n_in < 65) return;

  const int Nb = 4;
  char* ws = (char*)d_ws;
  size_t off = 0;
  auto alloc = [&](size_t bytes) -> void* {
    void* p = ws + off;
    off += (bytes + 255) & ~(size_t)255;
    return p;
  };

  // Activation / scratch buffers
  float*    F0 = (float*)alloc((size_t)Nb * 48 * 48 * 48 * 64 * sizeof(float));
  float*    Fd = (float*)alloc((size_t)Nb * 12 * 12 * 12 * 64 * sizeof(float));
  uint16_t* X0 = (uint16_t*)alloc((size_t)Nb * 96 * 96 * 96 * 3 * 2);
  uint16_t* S0 = (uint16_t*)alloc((size_t)Nb * 48 * 48 * 48 * 64 * 2);
  uint16_t* P[4];
  for (int i = 0; i < 4; ++i)
    P[i] = (uint16_t*)alloc((size_t)Nb * 24 * 24 * 24 * 64 * 2);
  float* d_scale = (float*)alloc(2048 * sizeof(float));
  float* d_bias  = (float*)alloc(2048 * sizeof(float));

  auto walloc = [&](size_t elems) -> uint16_t* {
    return (uint16_t*)alloc(elems * 2);
  };

  auto run_conv = [&](const uint16_t* src, int Di, int Cin, int Cout, int kx,
                      int s, const float* wraw, float* dst) -> int {
    const int Do = (Di + s - 1) / s;
    const int K  = kx * kx * kx * Cin;
    const int KP = ((K + 31) / 32) * 32;
    int pt = (Do - 1) * s + kx - Di; if (pt < 0) pt = 0;
    const int pad = pt / 2;
    uint16_t* wt = walloc((size_t)Cout * KP);
    long long wtot = (long long)Cout * KP;
    pack_weight_kernel<<<dim3((unsigned)((wtot + 255) / 256)), 256, 0, stream>>>(
        wraw, wt, Cin, kx, K, KP, wtot);
    const int M = Nb * Do * Do * Do;
    dim3 grid((unsigned)((M + BM - 1) / BM), (unsigned)(Cout / BN));
    conv3d_wmma<<<grid, 256, 0, stream>>>(src, wt, dst, Nb, Di, Do, Cin, Cout,
                                          kx, s, pad, K, KP);
    return Do;
  };

  auto run_norm = [&](const float* x, int M, int C, int groups, const float* g,
                      const float* b, const uint16_t* res, uint16_t* out,
                      int act) {
    const int S = M / groups;
    stats_kernel<<<dim3((unsigned)C, (unsigned)groups), 256, 0, stream>>>(
        x, g, b, d_scale, d_bias, S, C);
    long long total = (long long)M * C;
    norm_act_kernel<<<dim3((unsigned)((total + 255) / 256)), 256, 0, stream>>>(
        x, d_scale, d_bias, res, out, total, C, groups > 1 ? S : 0, act);
  };

  // ---- input indices (JAX pytree: sorted dict keys) ----
  // 0:x 1:conv5_w 2:final_b 3:final_w 4..63:layers 64:stem_w
  struct BlkIdx { int w1, g1, b1, w2, g2, b2, wd, gd, bd; };
  const BlkIdx blocks[8] = {
      {10, 7, 4, 11, 8, 5, 12, 9, 6},          // L1B0
      {17, 15, 13, 18, 16, 14, -1, -1, -1},    // L1B1
      {25, 22, 19, 26, 23, 20, 27, 24, 21},    // L2B0
      {32, 30, 28, 33, 31, 29, -1, -1, -1},    // L2B1
      {40, 37, 34, 41, 38, 35, 42, 39, 36},    // L3B0
      {47, 45, 43, 48, 46, 44, -1, -1, -1},    // L3B1
      {55, 52, 49, 56, 53, 50, 57, 54, 51},    // L4B0
      {62, 60, 58, 63, 61, 59, -1, -1, -1},    // L4B1
  };
  auto F = [&](int i) { return (const float*)d_in[i]; };

  // ---- stem: conv(3->64, s2) -> IN -> ReLU -> maxpool ----
  {
    long long total = (long long)Nb * 3 * 96 * 96 * 96;
    pack_input_kernel<<<dim3((unsigned)((total + 255) / 256)), 256, 0, stream>>>(
        F(0), X0, Nb, 3, 96);
  }
  run_conv(X0, 96, 3, 64, 3, 2, F(64), F0);                       // 96 -> 48
  run_norm(F0, Nb * 48 * 48 * 48, 64, Nb, nullptr, nullptr, nullptr, S0, 1);
  {
    long long total = (long long)Nb * 24 * 24 * 24 * 64;
    maxpool2_kernel<<<dim3((unsigned)((total + 255) / 256)), 256, 0, stream>>>(
        S0, P[0], Nb, 48, 64);
  }

  // ---- residual layers ----
  uint16_t* cur = P[0];
  uint16_t* mid = P[1];
  uint16_t* rsb = P[2];
  uint16_t* nxt = P[3];
  int D = 24, C = 64;
  const int Cpl[4] = {64, 128, 256, 512};

  for (int l = 0; l < 4; ++l) {
    const int Cout = Cpl[l];
    for (int bi = 0; bi < 2; ++bi) {
      const BlkIdx& B = blocks[l * 2 + bi];
      const int s = (bi == 0) ? 2 : 1;
      const int Do = run_conv(cur, D, C, Cout, 3, s, F(B.w1), F0);
      const int M = Nb * Do * Do * Do;
      run_norm(F0, M, Cout, 1, F(B.g1), F(B.b1), nullptr, mid, 1);
      run_conv(mid, Do, Cout, Cout, 3, 1, F(B.w2), F0);
      const uint16_t* res;
      if (B.wd >= 0) {
        run_conv(cur, D, C, Cout, 1, s, F(B.wd), Fd);
        run_norm(Fd, M, Cout, 1, F(B.gd), F(B.bd), nullptr, rsb, 0);
        res = rsb;
      } else {
        res = cur;
      }
      run_norm(F0, M, Cout, 1, F(B.g2), F(B.b2), res, nxt, 1);
      uint16_t* t = cur; cur = nxt; nxt = t;   // rotate
      D = Do; C = Cout;
    }
  }

  // ---- conv5 (s3) -> IN -> GELU -> global max -> linear ----
  run_conv(cur, D, 512, 512, 3, 3, F(1), F0);                     // 2 -> 1
  run_norm(F0, Nb * 1, 512, Nb, nullptr, nullptr, nullptr, mid, 2);
  head_kernel<<<1, 256, 0, stream>>>(mid, F(3), F(2), (float*)d_out, Nb, 1,
                                     512, 40);
}
